// Nearest_classifier_Mean_Metric_learning_41205916238322
// MI455X (gfx1250) — matrix operand order repaired and checked by reading, not compiled, not run on hardware
//
#include <hip/hip_runtime.h>
#include <hip/hip_bf16.h>

typedef __attribute__((ext_vector_type(16))) _Float16 v16h;
typedef __attribute__((ext_vector_type(8)))  _Float16 v8h;
typedef __attribute__((ext_vector_type(8)))  float    v8f;

constexpr int N = 16384;   // queries
constexpr int M = 10000;   // classes
constexpr int F = 2048;    // feature dim
constexpr int R = 256;     // metric rank

// ---------------- fragment helpers (wave32, gfx1250 WMMA layouts) ----------------

// A fragment 16x32 f16: lane<16 row holds contiguous K = k0+{0..15};
// lane>=16 same row, K = k0+{16..31}.  (matches B fragment K order)
__device__ __forceinline__ v16h a_frag_from_f32(const float* __restrict__ row, int k0, int half) {
  const float* p = row + k0 + 16 * half;
  v16h r;
#pragma unroll
  for (int i = 0; i < 8; ++i) r[i] = (_Float16)p[i];
#pragma unroll
  for (int i = 0; i < 8; ++i) r[8 + i] = (_Float16)p[8 + i];
  return r;
}

__device__ __forceinline__ v16h a_frag_from_f16(const _Float16* __restrict__ row, int k0, int half) {
  const _Float16* p = row + k0 + 16 * half;
  v8h lo = *(const v8h*)p;
  v8h hi = *(const v8h*)(p + 8);
  v16h r;
#pragma unroll
  for (int i = 0; i < 8; ++i) { r[i] = lo[i]; r[8 + i] = hi[i]; }
  return r;
}

// B fragment 32x16 f16: lane<16 holds column N=lane with contiguous K = k0+{0..15};
// lane>=16 holds K = k0+{16..31}.  (same K order as A fragment)
__device__ __forceinline__ v16h b_frag_from_f32(const float* __restrict__ row, int k0, int half) {
  const float* p = row + k0 + 16 * half;
  v16h r;
#pragma unroll
  for (int i = 0; i < 16; ++i) r[i] = (_Float16)p[i];
  return r;
}

__device__ __forceinline__ v16h b_frag_from_f16(const _Float16* __restrict__ row, int k0, int half) {
  return *(const v16h*)(row + k0 + 16 * half);  // 32B aligned: k0 % 32 == 0
}

__device__ __forceinline__ v8f wmma_f16(v16h a, v16h b, v8f c) {
  // (neg_a, A, neg_b, B, c_mod, C, reuse_a, reuse_b)
  return __builtin_amdgcn_wmma_f32_16x16x32_f16(false, a, false, b, (short)0, c, false, false);
}

// ---------------- kernel 1/2: H = X @ W^T  (f32 in, f16 out, fp32 accum) ----------------
// One wave per 16x16 output tile. ROWS = N or M; all dims divide exactly -> EXEC all ones.
template <int ROWS>
__global__ __launch_bounds__(256) void proj_kernel(const float* __restrict__ X,
                                                   const float* __restrict__ W,
                                                   _Float16* __restrict__ H) {
  constexpr int TM = ROWS / 16;
  constexpr int TR = R / 16;
  int wave = (blockIdx.x * blockDim.x + threadIdx.x) >> 5;
  if (wave >= TM * TR) return;                 // uniform per wave (grids are exact)
  int tile_m = wave / TR;
  int tile_r = wave % TR;
  int lane = threadIdx.x & 31;
  int half = lane >> 4;
  int l16  = lane & 15;

  const float* arow = X + (size_t)(tile_m * 16 + l16) * F;  // row of X
  const float* brow = W + (size_t)(tile_r * 16 + l16) * F;  // row of W == column of W^T

  v8f acc = {};
#pragma unroll 4
  for (int k0 = 0; k0 < F; k0 += 32) {
    __builtin_prefetch(arow + k0 + 128, 0, 1);              // global_prefetch_b8
    v16h a = a_frag_from_f32(arow, k0, half);
    v16h b = b_frag_from_f32(brow, k0, half);
    acc = wmma_f16(a, b, acc);
  }

  // C/D layout: VGPR j -> row (j + 8*half), col l16
  _Float16* hp = H + (size_t)(tile_m * 16) * R + tile_r * 16 + l16;
#pragma unroll
  for (int j = 0; j < 8; ++j)
    hp[(size_t)(j + 8 * half) * R] = (_Float16)acc[j];
}

// ---------------- kernel 3/4: sq[i] = || H[i,:] ||^2  (one wave per row) ----------------
__global__ __launch_bounds__(256) void sqnorm_kernel(const _Float16* __restrict__ H,
                                                     float* __restrict__ sq, int rows) {
  int row = (blockIdx.x * blockDim.x + threadIdx.x) >> 5;
  if (row >= rows) return;
  int lane = threadIdx.x & 31;
  const _Float16* hr = H + (size_t)row * R;
  float s = 0.f;
#pragma unroll
  for (int i = 0; i < R / 32; ++i) {
    float v = (float)hr[lane + 32 * i];
    s += v * v;
  }
#pragma unroll
  for (int off = 16; off > 0; off >>= 1)
    s += __shfl_xor(s, off, 32);
  if (lane == 0) sq[row] = s;
}

// ---------------- kernel 5: d2 = sqx + sqc - 2*hx@hc^T ; logit = -0.5*d2 ----------------
// One wave per 64x16 tile: 4 A-fragments reuse 1 B-fragment per K-step (32 WMMAs/tile).
__global__ __launch_bounds__(256) void pairdist_kernel(const _Float16* __restrict__ HX,
                                                       const _Float16* __restrict__ HC,
                                                       const float* __restrict__ SQX,
                                                       const float* __restrict__ SQC,
                                                       float* __restrict__ d2,
                                                       float* __restrict__ logit) {
  constexpr int TN = N / 64;   // 256
  constexpr int TM = M / 16;   // 625
  int wave = (blockIdx.x * blockDim.x + threadIdx.x) >> 5;
  if (wave >= TN * TM) return;
  int tile_n = wave / TM;
  int tile_m = wave % TM;
  int lane = threadIdx.x & 31;
  int half = lane >> 4;
  int l16  = lane & 15;
  int n0 = tile_n * 64;
  int m0 = tile_m * 16;

  const _Float16* br = HC + (size_t)(m0 + l16) * R;   // column of hc^T == row of hc
  const _Float16* ar = HX + (size_t)(n0 + l16) * R;

  v8f acc[4] = {};
#pragma unroll
  for (int k0 = 0; k0 < R; k0 += 32) {
    v16h b = b_frag_from_f16(br, k0, half);
#pragma unroll
    for (int i = 0; i < 4; ++i) {
      v16h a = a_frag_from_f16(ar + (size_t)(16 * i) * R, k0, half);
      acc[i] = wmma_f16(a, b, acc[i]);
    }
  }

  float sqc = SQC[m0 + l16];
#pragma unroll
  for (int i = 0; i < 4; ++i) {
    int nbase = n0 + 16 * i + 8 * half;
#pragma unroll
    for (int j = 0; j < 8; ++j) {
      int n = nbase + j;
      float v = SQX[n] + sqc - 2.0f * acc[i][j];
      size_t idx = (size_t)n * M + (size_t)(m0 + l16);
      d2[idx] = v;            // lanes 0..15 / 16..31 each cover 16 contiguous floats
      logit[idx] = -0.5f * v;
    }
  }
}

// ---------------- launch ----------------
extern "C" void kernel_launch(void* const* d_in, const int* in_sizes, int n_in,
                              void* d_out, int out_size, void* d_ws, size_t ws_size,
                              hipStream_t stream) {
  const float* x = (const float*)d_in[0];
  const float* C = (const float*)d_in[1];
  const float* W = (const float*)d_in[2];

  char* ws = (char*)d_ws;
  _Float16* hx = (_Float16*)ws;                                   // N*R f16 = 8 MB
  _Float16* hc = (_Float16*)(ws + (size_t)N * R * sizeof(_Float16));  // M*R f16 = 5 MB
  float* sqx = (float*)(ws + (size_t)(N + M) * R * sizeof(_Float16));
  float* sqc = sqx + N;

  float* d2    = (float*)d_out;
  float* logit = d2 + (size_t)N * M;

  // exact grids: every launched wave is fully active (EXEC all ones for WMMA)
  proj_kernel<N><<<(N / 16) * (R / 16) / 8, 256, 0, stream>>>(x, W, hx);   // 2048 blocks
  proj_kernel<M><<<(M / 16) * (R / 16) / 8, 256, 0, stream>>>(C, W, hc);   // 1250 blocks
  sqnorm_kernel<<<N / 8, 256, 0, stream>>>(hx, sqx, N);                    // 2048 blocks
  sqnorm_kernel<<<M / 8, 256, 0, stream>>>(hc, sqc, M);                    // 1250 blocks
  pairdist_kernel<<<(N / 64) * (M / 16) / 8, 256, 0, stream>>>(hx, hc, sqx, sqc, d2, logit); // 20000 blocks
}
